// MolGNN_19353122635825
// MI455X (gfx1250) — matrix operand order repaired and checked by reading, not compiled, not run on hardware
//
#include <hip/hip_runtime.h>
#include <hip/hip_bf16.h>

typedef __attribute__((ext_vector_type(16))) _Float16 v16h;
typedef __attribute__((ext_vector_type(8)))  _Float16 v8h;
typedef __attribute__((ext_vector_type(8)))  float    v8f;

#define N_NODES 50000
#define N_EDGES 200000
#define N_GRAPH 2500
#define G_PAD   2512      // N_GRAPH padded to multiple of 16
#define HDIM    256
#define HEADS   4
#define LAYERS  4
#define OUTDIM  768
#define ETOT    (N_EDGES + N_NODES)

// ---------------------------------------------------------------- embeddings
__global__ void atom_embed_kernel(const int* __restrict__ x,
                                  const float* __restrict__ emb,
                                  float* __restrict__ h,
                                  _Float16* __restrict__ h16) {
    const int aoff[9] = {0,119,128,139,151,160,165,173,175};
    __shared__ int idx[9];
    int n = blockIdx.x;
    if (threadIdx.x < 9) idx[threadIdx.x] = x[n * 9 + threadIdx.x] + aoff[threadIdx.x];
    __syncthreads();
    int c = threadIdx.x;
    float v = 0.f;
    #pragma unroll
    for (int j = 0; j < 9; ++j) v += emb[(size_t)idx[j] * HDIM + c];
    h[(size_t)n * HDIM + c] = v;
    h16[(size_t)n * HDIM + c] = (_Float16)v;
}

__global__ void bond_embed_kernel(const int* __restrict__ ea,
                                  const int* __restrict__ ei,
                                  const float* __restrict__ emb,
                                  _Float16* __restrict__ ea16,
                                  float* __restrict__ loop_sum) {
    const int boff[3] = {0,22,28};
    __shared__ int idx[3];
    __shared__ int dsts;
    int e = blockIdx.x;
    if (threadIdx.x < 3) idx[threadIdx.x] = ea[e * 3 + threadIdx.x] + boff[threadIdx.x];
    if (threadIdx.x == 0) dsts = ei[N_EDGES + e];
    __syncthreads();
    int c = threadIdx.x;
    float v = 0.f;
    #pragma unroll
    for (int j = 0; j < 3; ++j) v += emb[(size_t)idx[j] * HDIM + c];
    ea16[(size_t)e * HDIM + c] = (_Float16)v;
    atomicAdd(&loop_sum[(size_t)dsts * HDIM + c], v);
}

__global__ void deg_kernel(const int* __restrict__ ei, float* __restrict__ deg) {
    int e = blockIdx.x * blockDim.x + threadIdx.x;
    if (e < N_EDGES) atomicAdd(&deg[ei[N_EDGES + e]], 1.0f);
}

__global__ void loop_fin_kernel(const float* __restrict__ loop_sum,
                                const float* __restrict__ deg,
                                _Float16* __restrict__ ea16_loop) {
    int idx = blockIdx.x * blockDim.x + threadIdx.x;
    if (idx >= N_NODES * HDIM) return;
    int n = idx >> 8;
    float v = loop_sum[idx] / fmaxf(deg[n], 1.0f);
    ea16_loop[idx] = (_Float16)v;
}

// ---------------------------------------------------------------- weight prep
__global__ void transcvt_kernel(const float* __restrict__ W, _Float16* __restrict__ out,
                                int K, int Ncols) {
    int idx = blockIdx.x * blockDim.x + threadIdx.x;
    if (idx >= K * Ncols) return;
    int n = idx / K, k = idx - n * K;
    out[idx] = (_Float16)W[(size_t)k * Ncols + n];
}

__global__ void cvt16_kernel(const float* __restrict__ in, _Float16* __restrict__ out, int n) {
    int idx = blockIdx.x * blockDim.x + threadIdx.x;
    if (idx < n) out[idx] = (_Float16)in[idx];
}

__global__ void relu_cvt16_kernel(const float* __restrict__ in, _Float16* __restrict__ out, int n) {
    int idx = blockIdx.x * blockDim.x + threadIdx.x;
    if (idx < n) out[idx] = (_Float16)fmaxf(in[idx], 0.0f);
}

// ---------------------------------------------------------------- WMMA GEMM
// C[M,Ncols] = A16[M,256] @ W (Wt16 stored transposed: [Ncols][256]) + bias
// One wave -> 16x64 tile (4 accumulators share one A fragment).
// All 4 B fragments are staged into registers before the WMMA group so the
// loads clause together and the 4 WMMAs issue back-to-back.
// M must be a multiple of 16; Ncols a multiple of 256 per grid.y block of 4 waves.
template <typename OutT>
__global__ void gemm_wmma_kernel(const _Float16* __restrict__ A,
                                 const _Float16* __restrict__ Wt,
                                 const float* __restrict__ bias,
                                 OutT* __restrict__ C,
                                 int Ncols) {
    const int K = 256;
    int lane = threadIdx.x & 31;
    int wid  = threadIdx.x >> 5;
    int rowBase = blockIdx.x * 16;
    int colBase = (blockIdx.y * 4 + wid) * 64;
    int m    = lane & 15;
    int half = lane >> 4;

    const _Float16* ap  = A  + (size_t)(rowBase + m) * K + half * 8;
    const _Float16* bp0 = Wt + (size_t)(colBase + m) * K + half * 8;

    v8f acc[4] = {};
    #pragma unroll
    for (int kk = 0; kk < K; kk += 32) {
        v8h alo = *(const v8h*)(ap + kk);
        v8h ahi = *(const v8h*)(ap + kk + 16);
        v8h blo[4], bhi[4];
        #pragma unroll
        for (int t = 0; t < 4; ++t) {
            const _Float16* bp = bp0 + (size_t)t * 16 * K;
            blo[t] = *(const v8h*)(bp + kk);
            bhi[t] = *(const v8h*)(bp + kk + 16);
        }
        v16h a = __builtin_shufflevector(alo, ahi,
                                         0,1,2,3,4,5,6,7,8,9,10,11,12,13,14,15);
        #pragma unroll
        for (int t = 0; t < 4; ++t) {
            v16h b = __builtin_shufflevector(blo[t], bhi[t],
                                             0,1,2,3,4,5,6,7,8,9,10,11,12,13,14,15);
            acc[t] = __builtin_amdgcn_wmma_f32_16x16x32_f16(false, a, false, b,
                                                            (short)0, acc[t], false, false);
        }
    }

    #pragma unroll
    for (int t = 0; t < 4; ++t) {
        int col = colBase + t * 16 + m;
        float bv = bias ? bias[col] : 0.0f;
        #pragma unroll
        for (int r = 0; r < 8; ++r) {
            int orow = rowBase + r + half * 8;
            C[(size_t)orow * Ncols + col] = (OutT)(acc[t][r] + bv);
        }
    }
}

// ---------------------------------------------------------------- attention
__global__ void attn_logits_kernel(const float* __restrict__ xl,
                                   const float* __restrict__ xr,
                                   const _Float16* __restrict__ ee16,
                                   const int* __restrict__ ei,
                                   const float* __restrict__ attw,
                                   float* __restrict__ logits) {
    __shared__ float s_att[HDIM];
    for (int i = threadIdx.x; i < HDIM; i += blockDim.x) s_att[i] = attw[i];
    __syncthreads();
    int lane = threadIdx.x & 31;
    int wid  = threadIdx.x >> 5;
    int a = blockIdx.x * 4 + wid;
    if (a >= ETOT) return;
    int s = (a < N_EDGES) ? ei[a]           : (a - N_EDGES);
    int d = (a < N_EDGES) ? ei[N_EDGES + a] : (a - N_EDGES);
    int c0 = lane * 8;
    const float* pl = xl + (size_t)s * HDIM + c0;
    const float* pr = xr + (size_t)d * HDIM + c0;
    const _Float16* pe = ee16 + (size_t)a * HDIM + c0;
    float p = 0.f;
    #pragma unroll
    for (int j = 0; j < 8; ++j) {
        float v = pl[j] + pr[j] + (float)pe[j];
        v = (v > 0.f) ? v : 0.2f * v;
        p += v * s_att[c0 + j];
    }
    p += __shfl_xor(p, 1);
    p += __shfl_xor(p, 2);
    p += __shfl_xor(p, 4);
    if ((lane & 7) == 0) logits[(size_t)a * HEADS + (lane >> 3)] = p;
}

__device__ __forceinline__ unsigned enc_ord(float f) {
    unsigned u = __float_as_uint(f);
    return (u & 0x80000000u) ? ~u : (u | 0x80000000u);
}
__device__ __forceinline__ float dec_ord(unsigned k) {
    unsigned u = (k & 0x80000000u) ? (k & 0x7FFFFFFFu) : ~k;
    return __uint_as_float(u);
}

__global__ void attn_max_kernel(const float* __restrict__ logits,
                                const int* __restrict__ ei,
                                unsigned* __restrict__ lmaxenc) {
    int idx = blockIdx.x * blockDim.x + threadIdx.x;
    if (idx >= ETOT * HEADS) return;
    int a = idx >> 2, hh = idx & 3;
    int d = (a < N_EDGES) ? ei[N_EDGES + a] : (a - N_EDGES);
    atomicMax(&lmaxenc[d * HEADS + hh], enc_ord(logits[idx]));
}

__global__ void attn_expsum_kernel(const float* __restrict__ logits,
                                   const int* __restrict__ ei,
                                   const unsigned* __restrict__ lmaxenc,
                                   float* __restrict__ exv,
                                   float* __restrict__ denom) {
    int idx = blockIdx.x * blockDim.x + threadIdx.x;
    if (idx >= ETOT * HEADS) return;
    int a = idx >> 2, hh = idx & 3;
    int d = (a < N_EDGES) ? ei[N_EDGES + a] : (a - N_EDGES);
    float ex = expf(logits[idx] - dec_ord(lmaxenc[d * HEADS + hh]));
    exv[idx] = ex;
    atomicAdd(&denom[d * HEADS + hh], ex);
}

__global__ void attn_scatter_kernel(const float* __restrict__ xl,
                                    const int* __restrict__ ei,
                                    const float* __restrict__ exv,
                                    const float* __restrict__ denom,
                                    float* __restrict__ acc) {
    int lane = threadIdx.x & 31;
    int wid  = threadIdx.x >> 5;
    int a = blockIdx.x * 4 + wid;
    if (a >= ETOT) return;
    int s = (a < N_EDGES) ? ei[a]           : (a - N_EDGES);
    int d = (a < N_EDGES) ? ei[N_EDGES + a] : (a - N_EDGES);
    int hh = lane >> 3;
    float alpha = exv[(size_t)a * HEADS + hh] / denom[d * HEADS + hh];
    int c0 = lane * 8;
    const float* px = xl + (size_t)s * HDIM + c0;
    float* pd = acc + (size_t)d * HDIM + c0;
    #pragma unroll
    for (int j = 0; j < 8; ++j) atomicAdd(&pd[j], px[j] * alpha);
}

__global__ void node_update_kernel(const float* __restrict__ acc,
                                   const float* __restrict__ gbias,
                                   const float* __restrict__ gamma,
                                   const float* __restrict__ beta,
                                   const float* __restrict__ mean,
                                   const float* __restrict__ var,
                                   float* __restrict__ h,
                                   _Float16* __restrict__ h16) {
    int idx = blockIdx.x * blockDim.x + threadIdx.x;
    if (idx >= N_NODES * HDIM) return;
    int c = idx & 255;
    float v = acc[idx] + gbias[c];
    v = (v - mean[c]) * (gamma[c] * rsqrtf(var[c] + 1e-5f)) + beta[c];
    float hv = h[idx] + fmaxf(v, 0.0f);
    h[idx] = hv;
    h16[idx] = (_Float16)hv;
}

// ---------------------------------------------------------------- head
__global__ void pool_kernel(const float* __restrict__ h, const int* __restrict__ batch,
                            float* __restrict__ g) {
    int idx = blockIdx.x * blockDim.x + threadIdx.x;
    if (idx >= N_NODES * HDIM) return;
    int n = idx >> 8, c = idx & 255;
    atomicAdd(&g[(size_t)batch[n] * HDIM + c], h[idx]);
}

__global__ void norm_kernel(const float* __restrict__ in, float* __restrict__ out) {
    __shared__ float red[256];
    int gidx = blockIdx.x;
    const float* row = in + (size_t)gidx * OUTDIM;
    float s = 0.f;
    for (int i = threadIdx.x; i < OUTDIM; i += 256) { float v = row[i]; s += v * v; }
    red[threadIdx.x] = s;
    __syncthreads();
    for (int st = 128; st > 0; st >>= 1) {
        if (threadIdx.x < st) red[threadIdx.x] += red[threadIdx.x + st];
        __syncthreads();
    }
    float inv = 1.0f / fmaxf(sqrtf(red[0]), 1e-12f);
    for (int i = threadIdx.x; i < OUTDIM; i += 256)
        out[(size_t)gidx * OUTDIM + i] = row[i] * inv;
}

// ---------------------------------------------------------------- launch
extern "C" void kernel_launch(void* const* d_in, const int* in_sizes, int n_in,
                              void* d_out, int out_size, void* d_ws, size_t ws_size,
                              hipStream_t stream) {
    const int*   x         = (const int*)  d_in[0];
    const int*   edge_attr = (const int*)  d_in[1];
    const int*   edge_idx  = (const int*)  d_in[2];
    const int*   batch     = (const int*)  d_in[3];
    const float* atom_emb  = (const float*)d_in[5];
    const float* bond_emb  = (const float*)d_in[6];
    const float* Wl        = (const float*)d_in[7];
    const float* bl        = (const float*)d_in[8];
    const float* Wr        = (const float*)d_in[9];
    const float* br        = (const float*)d_in[10];
    const float* We        = (const float*)d_in[11];
    const float* att       = (const float*)d_in[12];
    const float* gat_bias  = (const float*)d_in[13];
    const float* bn_gamma  = (const float*)d_in[14];
    const float* bn_beta   = (const float*)d_in[15];
    const float* bn_mean   = (const float*)d_in[16];
    const float* bn_var    = (const float*)d_in[17];
    const float* proj_w1   = (const float*)d_in[18];
    const float* proj_b1   = (const float*)d_in[19];
    const float* proj_w2   = (const float*)d_in[20];
    const float* proj_b2   = (const float*)d_in[21];
    float* outp = (float*)d_out;

    // workspace layout
    char* base = (char*)d_ws;
    size_t off = 0;
    auto alloc = [&](size_t bytes) -> char* {
        char* p = base + off;
        off += (bytes + 255) & ~(size_t)255;
        return p;
    };
    const size_t NH = (size_t)N_NODES * HDIM, EH = (size_t)ETOT * HDIM;
    float*    h_buf  = (float*)   alloc(NH * 4);
    _Float16* h16    = (_Float16*)alloc(NH * 2);
    _Float16* ea16   = (_Float16*)alloc(EH * 2);
    _Float16* ee16   = (_Float16*)alloc(EH * 2);
    float*    xl     = (float*)   alloc(NH * 4);
    float*    xr     = (float*)   alloc(NH * 4);
    float*    acc    = (float*)   alloc(NH * 4);   // loop_sum, then per-layer accumulator
    float*    deg    = (float*)   alloc((size_t)N_NODES * 4);
    float*    logits = (float*)   alloc((size_t)ETOT * HEADS * 4);
    float*    exv    = (float*)   alloc((size_t)ETOT * HEADS * 4);
    unsigned* lmaxE  = (unsigned*)alloc((size_t)N_NODES * HEADS * 4);
    float*    denom  = (float*)   alloc((size_t)N_NODES * HEADS * 4);
    _Float16* WtL    = (_Float16*)alloc((size_t)LAYERS * HDIM * HDIM * 2);
    _Float16* WtR    = (_Float16*)alloc((size_t)LAYERS * HDIM * HDIM * 2);
    _Float16* WtE    = (_Float16*)alloc((size_t)LAYERS * HDIM * HDIM * 2);
    _Float16* pw1t   = (_Float16*)alloc((size_t)HDIM * HDIM * 2);
    _Float16* pw2t   = (_Float16*)alloc((size_t)OUTDIM * HDIM * 2);
    float*    gpool  = (float*)   alloc((size_t)G_PAD * HDIM * 4);
    _Float16* g16    = (_Float16*)alloc((size_t)G_PAD * HDIM * 2);
    float*    t1     = (float*)   alloc((size_t)G_PAD * HDIM * 4);
    _Float16* t116   = (_Float16*)alloc((size_t)G_PAD * HDIM * 2);
    float*    out2   = (float*)   alloc((size_t)G_PAD * OUTDIM * 4);
    (void)ws_size; (void)n_in; (void)in_sizes; (void)out_size;

    // ---- encoders + self-loop mean edge feature
    hipMemsetAsync(acc, 0, NH * 4, stream);
    hipMemsetAsync(deg, 0, (size_t)N_NODES * 4, stream);
    atom_embed_kernel<<<N_NODES, HDIM, 0, stream>>>(x, atom_emb, h_buf, h16);
    bond_embed_kernel<<<N_EDGES, HDIM, 0, stream>>>(edge_attr, edge_idx, bond_emb, ea16, acc);
    deg_kernel<<<(N_EDGES + 255) / 256, 256, 0, stream>>>(edge_idx, deg);
    loop_fin_kernel<<<(int)((NH + 255) / 256), 256, 0, stream>>>(
        acc, deg, ea16 + (size_t)N_EDGES * HDIM);

    // ---- weight prep (f16, transposed)
    const int WEL = HDIM * HDIM;
    for (int l = 0; l < LAYERS; ++l) {
        transcvt_kernel<<<(WEL + 255) / 256, 256, 0, stream>>>(Wl + (size_t)l * WEL, WtL + (size_t)l * WEL, HDIM, HDIM);
        transcvt_kernel<<<(WEL + 255) / 256, 256, 0, stream>>>(Wr + (size_t)l * WEL, WtR + (size_t)l * WEL, HDIM, HDIM);
        transcvt_kernel<<<(WEL + 255) / 256, 256, 0, stream>>>(We + (size_t)l * WEL, WtE + (size_t)l * WEL, HDIM, HDIM);
    }
    transcvt_kernel<<<(WEL + 255) / 256, 256, 0, stream>>>(proj_w1, pw1t, HDIM, HDIM);
    transcvt_kernel<<<(HDIM * OUTDIM + 255) / 256, 256, 0, stream>>>(proj_w2, pw2t, HDIM, OUTDIM);

    // ---- GNN layers (each block of 4 waves covers 16 rows x 256 cols)
    dim3 gbN(N_NODES / 16, HDIM / 256);
    dim3 gbE((ETOT + 15) / 16, HDIM / 256);
    for (int l = 0; l < LAYERS; ++l) {
        gemm_wmma_kernel<float><<<gbN, 128, 0, stream>>>(h16, WtL + (size_t)l * WEL, bl + l * HDIM, xl, HDIM);
        gemm_wmma_kernel<float><<<gbN, 128, 0, stream>>>(h16, WtR + (size_t)l * WEL, br + l * HDIM, xr, HDIM);
        gemm_wmma_kernel<_Float16><<<gbE, 128, 0, stream>>>(ea16, WtE + (size_t)l * WEL, nullptr, ee16, HDIM);

        hipMemsetAsync(lmaxE, 0, (size_t)N_NODES * HEADS * 4, stream);
        hipMemsetAsync(denom, 0, (size_t)N_NODES * HEADS * 4, stream);
        hipMemsetAsync(acc,   0, NH * 4, stream);

        attn_logits_kernel<<<(ETOT + 3) / 4, 128, 0, stream>>>(xl, xr, ee16, edge_idx, att + l * HDIM, logits);
        attn_max_kernel<<<(ETOT * HEADS + 255) / 256, 256, 0, stream>>>(logits, edge_idx, lmaxE);
        attn_expsum_kernel<<<(ETOT * HEADS + 255) / 256, 256, 0, stream>>>(logits, edge_idx, lmaxE, exv, denom);
        attn_scatter_kernel<<<(ETOT + 3) / 4, 128, 0, stream>>>(xl, edge_idx, exv, denom, acc);
        node_update_kernel<<<(int)((NH + 255) / 256), 256, 0, stream>>>(
            acc, gat_bias + l * HDIM, bn_gamma + l * HDIM, bn_beta + l * HDIM,
            bn_mean + l * HDIM, bn_var + l * HDIM, h_buf, h16);
    }

    // ---- readout head (M padded to G_PAD; pad rows of g16 are zero)
    hipMemsetAsync(gpool, 0, (size_t)G_PAD * HDIM * 4, stream);
    hipMemsetAsync(g16,   0, (size_t)G_PAD * HDIM * 2, stream);
    pool_kernel<<<(int)((NH + 255) / 256), 256, 0, stream>>>(h_buf, batch, gpool);
    cvt16_kernel<<<(N_GRAPH * HDIM + 255) / 256, 256, 0, stream>>>(gpool, g16, N_GRAPH * HDIM);
    dim3 gbG1(G_PAD / 16, HDIM / 256);
    gemm_wmma_kernel<float><<<gbG1, 128, 0, stream>>>(g16, pw1t, proj_b1, t1, HDIM);
    relu_cvt16_kernel<<<(G_PAD * HDIM + 255) / 256, 256, 0, stream>>>(t1, t116, G_PAD * HDIM);
    dim3 gbG2(G_PAD / 16, OUTDIM / 256);
    gemm_wmma_kernel<float><<<gbG2, 128, 0, stream>>>(t116, pw2t, proj_b2, out2, OUTDIM);
    norm_kernel<<<N_GRAPH, 256, 0, stream>>>(out2, outp);
}
